// ProgramLearner_50199577756094
// MI455X (gfx1250) — matrix operand order repaired and checked by reading, not compiled, not run on hardware
//
#include <hip/hip_runtime.h>
#include <hip/hip_bf16.h>

// ---- problem constants (from reference) ----
#define BB   4
#define NN   20000
#define MM   64
#define WLEN 8
#define TT   2

#define NT     32            // n-values per block -> 128 (b,n) columns
#define COLS   (NT * BB)     // 128
#define CSTR   80            // bf16 elements per column in LDS (160B, 32B-aligned)

typedef __attribute__((ext_vector_type(16))) __bf16 v16bf;
typedef __attribute__((ext_vector_type(8)))  float  v8f;

__device__ __forceinline__ unsigned short f32_to_bf16(float f) {
    unsigned int u = __builtin_bit_cast(unsigned int, f);
    unsigned int r = u + 0x7FFFu + ((u >> 16) & 1u);   // round-to-nearest-even
    return (unsigned short)(r >> 16);
}
__device__ __forceinline__ float bf16_to_f32(unsigned int h) {
    return __builtin_bit_cast(float, h << 16);
}

// -------- kernel 1: pi = softmax(W.flatten()).reshape(64,64), stored bf16 row-major --------
__global__ void pl_softmax_pi(const float* __restrict__ Wm, unsigned short* __restrict__ piB) {
    __shared__ float red[256];
    const int tid = threadIdx.x;
    float lmax = -3.402823466e38f;
    for (int i = tid; i < MM * MM; i += 256) lmax = fmaxf(lmax, Wm[i]);
    red[tid] = lmax; __syncthreads();
    for (int s = 128; s > 0; s >>= 1) {
        if (tid < s) red[tid] = fmaxf(red[tid], red[tid + s]);
        __syncthreads();
    }
    const float gmax = red[0]; __syncthreads();
    float lsum = 0.f;
    for (int i = tid; i < MM * MM; i += 256) lsum += expf(Wm[i] - gmax);
    red[tid] = lsum; __syncthreads();
    for (int s = 128; s > 0; s >>= 1) {
        if (tid < s) red[tid] += red[tid + s];
        __syncthreads();
    }
    const float inv = 1.0f / red[0];
    for (int i = tid; i < MM * MM; i += 256)
        piB[i] = f32_to_bf16(expf(Wm[i] - gmax) * inv);
}

// -------- kernel 2: aT[n][b] = a[b][n]  (float4 per n so one gather serves all 4 batches) --------
__global__ void pl_transpose_a(const float* __restrict__ a, float* __restrict__ aT) {
    const int n = blockIdx.x * blockDim.x + threadIdx.x;
    if (n < NN) {
        float4 v;
        v.x = a[0 * NN + n];
        v.y = a[1 * NN + n];
        v.z = a[2 * NN + n];
        v.w = a[3 * NN + n];
        ((float4*)aT)[n] = v;
    }
}

__device__ __forceinline__ void acc_pair(const float4* __restrict__ aT4, int i0, int i1, float4& m) {
    const float4 u = aT4[i0];
    const float4 v = aT4[i1];
    m.x = fmaxf(m.x, u.x * v.x);
    m.y = fmaxf(m.y, u.y * v.y);
    m.z = fmaxf(m.z, u.z * v.z);
    m.w = fmaxf(m.w, u.w * v.w);
}

// -------- kernel 3: fused gather + WMMA quadratic form --------
// derived[b,n] = 1 - G1(:,c)^T * pi * G2(:,c),  G = 1 - max_w a[x0]*a[x1]
// a_next[b,n]  = 1 - (1 - a[b,n]) * (G1^T pi G2)
__global__ __launch_bounds__(256)
void pl_main(const float* __restrict__ aT,
             const unsigned short* __restrict__ piB,
             const int* __restrict__ X1,
             const int* __restrict__ X2,
             float* __restrict__ a_next) {
    __shared__ unsigned short G1[COLS * CSTR];   // 20480 B, bf16, column-major (j contiguous)
    __shared__ unsigned short G2[COLS * CSTR];   // 20480 B

    const int tid = threadIdx.x;
    const int n0  = blockIdx.x * NT;
    const float4* __restrict__ aT4 = (const float4*)aT;

    // ---- phase 1: gather F, write G = 1 - F as bf16 into LDS ----
    // tasks: sel(2) x j(64) x nn(32) = 4096; 16 per thread; within a wave j/nn give
    // fully-coalesced 64B index reads per lane.
    for (int t = tid; t < 2 * MM * NT; t += 256) {
        const int  tp  = t & (MM * NT - 1);
        const bool one = (t < MM * NT);
        const int* __restrict__ X = one ? X1 : X2;
        unsigned short* __restrict__ G = one ? G1 : G2;
        const int j  = tp >> 5;        // 0..63
        const int nn = tp & (NT - 1);  // 0..31
        const int n  = n0 + nn;
        if (n < NN) {
            const int4* __restrict__ xp = (const int4*)(X + ((size_t)j * NN + n) * (WLEN * 2));
            // prefetch the index block this thread touches 8 j-steps later (gfx1250 global_prefetch)
            if (j + 8 < MM)
                __builtin_prefetch((const void*)(xp + (size_t)8 * NN * 4), 0, 1);
            const int4 q0 = xp[0], q1 = xp[1], q2 = xp[2], q3 = xp[3];
            float4 m = make_float4(-3.402823466e38f, -3.402823466e38f,
                                   -3.402823466e38f, -3.402823466e38f);
            acc_pair(aT4, q0.x, q0.y, m);
            acc_pair(aT4, q0.z, q0.w, m);
            acc_pair(aT4, q1.x, q1.y, m);
            acc_pair(aT4, q1.z, q1.w, m);
            acc_pair(aT4, q2.x, q2.y, m);
            acc_pair(aT4, q2.z, q2.w, m);
            acc_pair(aT4, q3.x, q3.y, m);
            acc_pair(aT4, q3.z, q3.w, m);
            const int cbase = nn * BB;
            G[(cbase + 0) * CSTR + j] = f32_to_bf16(1.0f - m.x);
            G[(cbase + 1) * CSTR + j] = f32_to_bf16(1.0f - m.y);
            G[(cbase + 2) * CSTR + j] = f32_to_bf16(1.0f - m.z);
            G[(cbase + 3) * CSTR + j] = f32_to_bf16(1.0f - m.w);
        } else {
            const int cbase = nn * BB;
            for (int b = 0; b < BB; ++b) G[(cbase + b) * CSTR + j] = 0;
        }
    }
    __syncthreads();

    // ---- phase 2: s(c) = G1(:,c)^T * pi * G2(:,c) via v_wmma_f32_16x16x32_bf16 ----
    const int lane    = tid & 31;
    const int cg      = tid >> 5;            // 8 waves -> 8 col-groups of 16
    const int colbase = cg * 16;
    const int l16     = lane & 15;
    const int hi      = (lane >= 16) ? 1 : 0;
    float s = 0.0f;

    for (int rb = 0; rb < 4; ++rb) {
        v8f acc = {};
        #pragma unroll
        for (int kc = 0; kc < 2; ++kc) {
            // A fragment: pi rows rb*16..+15, K = kc*32..+31 (16-bit A layout)
            const int row = rb * 16 + l16;
            const int e0  = kc * 32 + (hi ? 8 : 0);        // first element offset in row
            const uint4* prow = (const uint4*)(piB + row * MM);
            union { uint4 q[2]; v16bf v; } Af;
            Af.q[0] = prow[e0 >> 3];
            Af.q[1] = prow[(e0 >> 3) + 2];
            // B fragment: G2[K = kc*32 + (hi?8:0) + {0..7, 16..23}][col] (j-contiguous in LDS)
            const uint4* bcol = (const uint4*)(&G2[(colbase + l16) * CSTR + kc * 32 + hi * 8]);
            union { uint4 q[2]; v16bf v; } Bf;
            Bf.q[0] = bcol[0];
            Bf.q[1] = bcol[2];
            acc = __builtin_amdgcn_wmma_f32_16x16x32_bf16(
                      false, Af.v, false, Bf.v, (short)0, acc, false, false);
        }
        // diag accumulate: s += sum_i G1[j(i), col] * acc[i]
        // C layout: VGPR i <-> row rb*16 + i (+8 for upper half-lanes)
        const int jbase = rb * 16 + hi * 8;
        const uint4 g = *(const uint4*)(&G1[(colbase + l16) * CSTR + jbase]);
        const unsigned int gw[4] = {g.x, g.y, g.z, g.w};
        #pragma unroll
        for (int i = 0; i < 4; ++i) {
            s += bf16_to_f32(gw[i] & 0xFFFFu) * acc[2 * i + 0];
            s += bf16_to_f32(gw[i] >> 16)     * acc[2 * i + 1];
        }
    }
    // lanes l and l+16 hold complementary row-halves of the same column
    s += __shfl_xor(s, 16, 32);

    if (lane < 16) {
        const int c = colbase + lane;
        const int n = n0 + (c >> 2);
        const int b = c & 3;
        if (n < NN) {
            const float aold = aT[n * 4 + b];
            a_next[b * NN + n] = 1.0f - (1.0f - aold) * s;
        }
    }
}

extern "C" void kernel_launch(void* const* d_in, const int* in_sizes, int n_in,
                              void* d_out, int out_size, void* d_ws, size_t ws_size,
                              hipStream_t stream) {
    const float* a0 = (const float*)d_in[0];
    const float* Wm = (const float*)d_in[1];
    const int*   X1 = (const int*)d_in[2];
    const int*   X2 = (const int*)d_in[3];
    float* out = (float*)d_out;

    char* ws = (char*)d_ws;
    unsigned short* piB = (unsigned short*)ws;                    // 8192 B
    float* aT   = (float*)(ws + 8192);                            // N*4 floats = 320000 B
    float* aMid = (float*)(ws + 8192 + (size_t)NN * 4 * sizeof(float));

    const int tblocks = (NN + 255) / 256;
    const int mblocks = (NN + NT - 1) / NT;                       // 625

    // pi is constant across iterations
    pl_softmax_pi<<<1, 256, 0, stream>>>(Wm, piB);

    // iteration 0: a0 -> aMid
    pl_transpose_a<<<tblocks, 256, 0, stream>>>(a0, aT);
    pl_main<<<mblocks, 256, 0, stream>>>(aT, piB, X1, X2, aMid);

    // iteration 1: aMid -> out
    pl_transpose_a<<<tblocks, 256, 0, stream>>>(aMid, aT);
    pl_main<<<mblocks, 256, 0, stream>>>(aT, piB, X1, X2, out);
}
